// SelfAttention_86835648790793
// MI455X (gfx1250) — matrix operand order repaired and checked by reading, not compiled, not run on hardware
//
#include <hip/hip_runtime.h>

// ---------------------------------------------------------------------------
// Types / fragment helpers
// ---------------------------------------------------------------------------
typedef __attribute__((ext_vector_type(16))) __bf16 v16bf;
typedef __attribute__((ext_vector_type(8)))  float  v8f;
typedef __attribute__((ext_vector_type(4)))  unsigned int u32x4;

union Frag32 { v16bf bf; u32x4 u[2]; };
union Pack16 { u32x4 u; __bf16 h[8]; };

static __device__ inline v16bf frag_split(const __bf16* p0, const __bf16* p1) {
  Frag32 f;
  f.u[0] = *(const u32x4*)p0;
  f.u[1] = *(const u32x4*)p1;
  return f.bf;
}
static __device__ inline v16bf frag_contig(const __bf16* p) {
  Frag32 f;
  f.u[0] = ((const u32x4*)p)[0];
  f.u[1] = ((const u32x4*)p)[1];
  return f.bf;
}
static __device__ inline v8f zero8() {
  v8f z = {0.f, 0.f, 0.f, 0.f, 0.f, 0.f, 0.f, 0.f};
  return z;
}

// ---------------------------------------------------------------------------
// CDNA5 async global->LDS copy (16B per lane, ASYNCcnt-tracked).
// Generic pointers to __shared__ carry the LDS byte offset in the low 32 bits.
// ---------------------------------------------------------------------------
static __device__ inline unsigned lds_off(const void* p) {
  return (unsigned)(unsigned long long)(uintptr_t)p;
}
static __device__ inline void async_b128(void* lds_dst, const void* gsrc) {
  unsigned l = lds_off(lds_dst);
  unsigned long long g = (unsigned long long)(uintptr_t)gsrc;
  asm volatile("global_load_async_to_lds_b128 %0, %1, off"
               :: "v"(l), "v"(g) : "memory");
}
static __device__ inline void wait_async() {
  asm volatile("s_wait_asynccnt 0x0" ::: "memory");
}

// ---------------------------------------------------------------------------
// fp32 -> bf16 conversion
// ---------------------------------------------------------------------------
__global__ void cvt_f32_bf16(const float* __restrict__ src,
                             __bf16* __restrict__ dst, int n) {
  int i = blockIdx.x * blockDim.x + threadIdx.x;
  if (i < n) dst[i] = (__bf16)src[i];
}

// ---------------------------------------------------------------------------
// 64x64-tile GEMM: C[M,N] = scale * (A[M,K] @ B[K,N]); bf16 in, bf16/f32 out
// 128 threads = 4 waves; each wave computes a 16x64 strip via 4 WMMA accums.
// A tile staged with async global->LDS DMA; B tile transposed through VGPRs.
// ---------------------------------------------------------------------------
template <bool F32OUT>
__global__ __launch_bounds__(128) void gemm_bf16_64x64(
    const __bf16* __restrict__ A, const __bf16* __restrict__ B,
    void* __restrict__ Cout, int M, int N, int K, float scale) {
  __shared__ __attribute__((aligned(16))) __bf16 As[64][40];  // row-major [m][k]
  __shared__ __attribute__((aligned(16))) __bf16 Bt[64][40];  // transposed [n][k]

  const int tid  = threadIdx.x;
  const int wave = tid >> 5;
  const int lane = tid & 31;
  const int hsel = lane >> 4;  // which half of the wave
  const int ln   = lane & 15;
  const int mb   = blockIdx.x * 64;
  const int nb   = blockIdx.y * 64;

  v8f acc[4];
#pragma unroll
  for (int ct = 0; ct < 4; ++ct) acc[ct] = zero8();

  for (int kk = 0; kk < K; kk += 32) {
    // stage A tile 64x32 via async LDS DMA (two b128 per thread)
    {
      int r = tid >> 1, sg = (tid & 1) * 16;
      const __bf16* g = &A[(size_t)(mb + r) * K + kk + sg];
      async_b128(&As[r][sg],     g);
      async_b128(&As[r][sg + 8], g + 8);
    }
    // stage B tile 32x64 transposed into Bt[n][k]
    {
      int k = tid >> 2, c0 = (tid & 3) * 16;
      const u32x4* g = (const u32x4*)&B[(size_t)(kk + k) * N + nb + c0];
      Pack16 p0, p1;
      p0.u = g[0];
      p1.u = g[1];
#pragma unroll
      for (int j = 0; j < 8; ++j) {
        Bt[c0 + j][k]     = p0.h[j];
        Bt[c0 + 8 + j][k] = p1.h[j];
      }
    }
    if (kk + 32 < K) {  // global_prefetch_b8 for the next tiles
      __builtin_prefetch(&A[(size_t)(mb + (tid >> 1)) * K + kk + 32], 0, 0);
      __builtin_prefetch(&B[(size_t)(kk + 32 + (tid >> 2)) * N + nb], 0, 0);
    }
    wait_async();
    __syncthreads();

    const int row = (wave << 4) + ln;
    // A frag: lanes 0-15 get K 0-15, lanes 16-31 get K 16-31
    v16bf a = frag_contig(&As[row][hsel * 16]);
#pragma unroll
    for (int ct = 0; ct < 4; ++ct) {
      v16bf b = frag_contig(&Bt[ct * 16 + ln][hsel * 16]);
      acc[ct] = __builtin_amdgcn_wmma_f32_16x16x32_bf16(
          false, a, false, b, (short)0, acc[ct], false, false);
    }
    __syncthreads();
  }

#pragma unroll
  for (int ct = 0; ct < 4; ++ct) {
#pragma unroll
    for (int i = 0; i < 8; ++i) {
      int gr = mb + (wave << 4) + i + hsel * 8;  // C/D layout: VGPR i -> row i / i+8
      int gc = nb + ct * 16 + ln;
      float v = acc[ct][i] * scale;
      if (F32OUT)
        ((float*)Cout)[(size_t)gr * N + gc] = v;
      else
        ((__bf16*)Cout)[(size_t)gr * N + gc] = (__bf16)v;
    }
  }
}

// ---------------------------------------------------------------------------
// Flash attention: one workgroup per (b, h, 64-row q tile). 4 waves x 16 rows.
// Q/K pre-scaled by d^-0.25. Streams causal 64-key blocks with online softmax.
// Q and K tiles staged with async global->LDS DMA; V transposed through VGPRs.
// ---------------------------------------------------------------------------
__global__ __launch_bounds__(128) void attn_fwd(
    const __bf16* __restrict__ Q, const __bf16* __restrict__ Kb,
    const __bf16* __restrict__ V, __bf16* __restrict__ O) {
  constexpr int T = 2048, NH = 16, D = 64, LDW = 72;  // padded LDS row stride
  __shared__ __attribute__((aligned(16))) __bf16 Qs[64 * LDW];
  __shared__ __attribute__((aligned(16))) __bf16 Ks[64 * LDW];
  __shared__ __attribute__((aligned(16))) __bf16 Vt[64 * LDW];      // [d][key]
  __shared__ __attribute__((aligned(16))) __bf16 Ps[4][16 * LDW];   // per-wave P

  const int tid  = threadIdx.x;
  const int wave = tid >> 5;
  const int lane = tid & 31;
  const int hsel = lane >> 4;
  const int ln   = lane & 15;
  const int b    = blockIdx.z;
  const int h    = blockIdx.y;
  const int qb   = blockIdx.x * 64;
  const size_t rs   = (size_t)NH * D;                 // 1024 row stride
  const size_t base = ((size_t)b * T) * rs + (size_t)h * D;

  // stage Q tile once (async DMA; first in-loop wait/barrier covers it)
  {
    int r = tid >> 1, sg = (tid & 1) * 32;
    const __bf16* g = &Q[base + (size_t)(qb + r) * rs + sg];
#pragma unroll
    for (int j = 0; j < 4; ++j) async_b128(&Qs[r * LDW + sg + 8 * j], g + 8 * j);
  }

  float m_run[8], l_run[8];
  v8f o[4];
#pragma unroll
  for (int i = 0; i < 8; ++i) { m_run[i] = -1e30f; l_run[i] = 0.f; }
#pragma unroll
  for (int ct = 0; ct < 4; ++ct) o[ct] = zero8();

  const int qrow0 = qb + (wave << 4) + hsel * 8;  // + i gives this lane's rows

  for (int kb = 0; kb <= qb; kb += 64) {
    // stage K tile (async DMA)
    {
      int r = tid >> 1, sg = (tid & 1) * 32;
      const __bf16* g = &Kb[base + (size_t)(kb + r) * rs + sg];
#pragma unroll
      for (int j = 0; j < 4; ++j) async_b128(&Ks[r * LDW + sg + 8 * j], g + 8 * j);
    }
    // stage V tile transposed -> Vt[d][key]
    {
      int r = tid >> 1, sg = (tid & 1) * 32;
      const u32x4* g = (const u32x4*)&V[base + (size_t)(kb + r) * rs + sg];
#pragma unroll
      for (int j = 0; j < 4; ++j) {
        Pack16 p;
        p.u = g[j];
#pragma unroll
        for (int e = 0; e < 8; ++e) Vt[(sg + 8 * j + e) * LDW + r] = p.h[e];
      }
    }
    if (kb + 64 <= qb) {
      __builtin_prefetch(&Kb[base + (size_t)(kb + 64 + (tid >> 1)) * rs], 0, 0);
      __builtin_prefetch(&V[base + (size_t)(kb + 64 + (tid >> 1)) * rs], 0, 0);
    }
    wait_async();
    __syncthreads();

    // S = Q K^T : 16x64 scores per wave (8 WMMAs)
    const int row = (wave << 4) + ln;
    v16bf aq0 = frag_contig(&Qs[row * LDW + hsel * 16]);
    v16bf aq1 = frag_contig(&Qs[row * LDW + 32 + hsel * 16]);
    v8f s[4];
#pragma unroll
    for (int ct = 0; ct < 4; ++ct) {
      s[ct] = zero8();
      v16bf b0 = frag_contig(&Ks[(ct * 16 + ln) * LDW + hsel * 16]);
      s[ct] = __builtin_amdgcn_wmma_f32_16x16x32_bf16(false, aq0, false, b0,
                                                      (short)0, s[ct], false, false);
      v16bf b1 = frag_contig(&Ks[(ct * 16 + ln) * LDW + 32 + hsel * 16]);
      s[ct] = __builtin_amdgcn_wmma_f32_16x16x32_bf16(false, aq1, false, b1,
                                                      (short)0, s[ct], false, false);
    }

    // causal mask (strictly-upper -> -inf), only on the diagonal block
    if (kb == qb) {
#pragma unroll
      for (int ct = 0; ct < 4; ++ct) {
        int key = kb + ct * 16 + ln;
#pragma unroll
        for (int i = 0; i < 8; ++i)
          if (key > qrow0 + i) s[ct][i] = -1e30f;
      }
    }

    // online softmax; xor masks 1/2/4/8 reduce within each 16-lane half,
    // which is exactly one row set of the C-layout.
#pragma unroll
    for (int i = 0; i < 8; ++i) {
      float bm = fmaxf(fmaxf(s[0][i], s[1][i]), fmaxf(s[2][i], s[3][i]));
      bm = fmaxf(bm, __shfl_xor(bm, 1));
      bm = fmaxf(bm, __shfl_xor(bm, 2));
      bm = fmaxf(bm, __shfl_xor(bm, 4));
      bm = fmaxf(bm, __shfl_xor(bm, 8));
      float mn = fmaxf(m_run[i], bm);
      float sc = __expf(m_run[i] - mn);
      float bs = 0.f;
#pragma unroll
      for (int ct = 0; ct < 4; ++ct) {
        float p = __expf(s[ct][i] - mn);
        s[ct][i] = p;
        bs += p;
      }
      bs += __shfl_xor(bs, 1);
      bs += __shfl_xor(bs, 2);
      bs += __shfl_xor(bs, 4);
      bs += __shfl_xor(bs, 8);
      l_run[i] = l_run[i] * sc + bs;
      m_run[i] = mn;
#pragma unroll
      for (int ct = 0; ct < 4; ++ct) o[ct][i] *= sc;
    }

    // reshape P (C-layout f32) -> A-layout bf16 via per-wave LDS buffer
#pragma unroll
    for (int ct = 0; ct < 4; ++ct)
#pragma unroll
      for (int i = 0; i < 8; ++i)
        Ps[wave][(i + hsel * 8) * LDW + ct * 16 + ln] = (__bf16)s[ct][i];
    asm volatile("s_wait_dscnt 0" ::: "memory");  // same-wave LDS RAW

    // O += P V (8 WMMAs)
    v16bf ap0 = frag_contig(&Ps[wave][ln * LDW + hsel * 16]);
    v16bf ap1 = frag_contig(&Ps[wave][ln * LDW + 32 + hsel * 16]);
#pragma unroll
    for (int ct = 0; ct < 4; ++ct) {
      v16bf b0 = frag_contig(&Vt[(ct * 16 + ln) * LDW + hsel * 16]);
      o[ct] = __builtin_amdgcn_wmma_f32_16x16x32_bf16(false, ap0, false, b0,
                                                      (short)0, o[ct], false, false);
      v16bf b1 = frag_contig(&Vt[(ct * 16 + ln) * LDW + 32 + hsel * 16]);
      o[ct] = __builtin_amdgcn_wmma_f32_16x16x32_bf16(false, ap1, false, b1,
                                                      (short)0, o[ct], false, false);
    }
    __syncthreads();
  }

  // normalize and store attention output (bf16, [b*t, h*d] layout)
#pragma unroll
  for (int ct = 0; ct < 4; ++ct)
#pragma unroll
    for (int i = 0; i < 8; ++i) {
      int qr = qrow0 + i;
      float v = o[ct][i] / l_run[i];
      O[base + (size_t)qr * rs + ct * 16 + ln] = (__bf16)v;
    }
}

// ---------------------------------------------------------------------------
// Launch: x/W -> bf16, QKV GEMMs (Q,K pre-scaled by d^-0.25), flash attention,
// output projection into fp32 d_out.
// ---------------------------------------------------------------------------
extern "C" void kernel_launch(void* const* d_in, const int* in_sizes, int n_in,
                              void* d_out, int out_size, void* d_ws, size_t ws_size,
                              hipStream_t stream) {
  constexpr int B = 4, T = 2048, E = 1024;
  constexpr int M = B * T;           // 8192 rows
  constexpr int NX = M * E;          // 8388608 x / q / k / v / attn elems
  constexpr int NW = E * E;          // 1048576 weight elems
  const float QK_SCALE = 0.3535533906f;  // 64^-0.25

  const float* x  = (const float*)d_in[0];
  const float* Wq = (const float*)d_in[1];
  const float* Wk = (const float*)d_in[2];
  const float* Wv = (const float*)d_in[3];
  const float* Wo = (const float*)d_in[4];

  char* ws = (char*)d_ws;
  __bf16* Xb  = (__bf16*)(ws);
  __bf16* Wqb = (__bf16*)(ws + (size_t)NX * 2);
  __bf16* Wkb = Wqb + NW;
  __bf16* Wvb = Wkb + NW;
  __bf16* Wob = Wvb + NW;
  __bf16* Qb  = (__bf16*)((char*)(Wob + NW));
  __bf16* Kb  = Qb + NX;
  __bf16* Vb  = Kb + NX;
  __bf16* Ab  = Vb + NX;  // attention output, pre-projection

  cvt_f32_bf16<<<(NX + 255) / 256, 256, 0, stream>>>(x, Xb, NX);
  cvt_f32_bf16<<<(NW + 255) / 256, 256, 0, stream>>>(Wq, Wqb, NW);
  cvt_f32_bf16<<<(NW + 255) / 256, 256, 0, stream>>>(Wk, Wkb, NW);
  cvt_f32_bf16<<<(NW + 255) / 256, 256, 0, stream>>>(Wv, Wvb, NW);
  cvt_f32_bf16<<<(NW + 255) / 256, 256, 0, stream>>>(Wo, Wob, NW);

  dim3 ggrid(M / 64, E / 64);
  gemm_bf16_64x64<false><<<ggrid, 128, 0, stream>>>(Xb, Wqb, Qb, M, E, E, QK_SCALE);
  gemm_bf16_64x64<false><<<ggrid, 128, 0, stream>>>(Xb, Wkb, Kb, M, E, E, QK_SCALE);
  gemm_bf16_64x64<false><<<ggrid, 128, 0, stream>>>(Xb, Wvb, Vb, M, E, E, 1.0f);

  attn_fwd<<<dim3(T / 64, 16, B), 128, 0, stream>>>(Qb, Kb, Vb, Ab);

  gemm_bf16_64x64<true><<<ggrid, 128, 0, stream>>>(Ab, Wob, (float*)d_out, M, E, E, 1.0f);
}
